// SqueezeNet_57801669870044
// MI455X (gfx1250) — 8-bit matrix operand order repaired and checked by reading, compiled, not run on hardware
//
#include <hip/hip_runtime.h>
#include <stdint.h>

typedef __attribute__((ext_vector_type(8))) int v8i;

__device__ __forceinline__ void async_ld_b64(uint32_t lds, const void* g) {
  asm volatile("global_load_async_to_lds_b64 %0, %1, off" :: "v"(lds), "v"(g) : "memory");
}
__device__ __forceinline__ void async_ld_b128(uint32_t lds, const void* g) {
  asm volatile("global_load_async_to_lds_b128 %0, %1, off" :: "v"(lds), "v"(g) : "memory");
}
__device__ __forceinline__ void wait_async0() {
  asm volatile("s_wait_asynccnt 0x0" ::: "memory");
}

__global__ void repack_w_kernel(const float* __restrict__ w, int8_t* __restrict__ bt,
                                int KH, int KW, int Ci, int Co, int Kpad) {
  int k  = blockIdx.x * blockDim.x + threadIdx.x;
  int co = blockIdx.y;
  if (k >= Kpad) return;
  int K = KH * KW * Ci;
  int8_t v = 0;
  if (k < K) v = (w[(size_t)k * Co + co] >= 0.0f) ? (int8_t)1 : (int8_t)-1;
  bt[(size_t)co * Kpad + k] = v;
}

__global__ void stem_conv_kernel(const float* __restrict__ xin, const float* __restrict__ w0,
                                 const float* __restrict__ b0, const float* __restrict__ bn0,
                                 int8_t* __restrict__ out) {
  constexpr int H = 224, W = 224;
  int tx = blockIdx.x * blockDim.x + threadIdx.x;
  int co = tx & 31;
  int ox = tx >> 5;
  int noy = blockIdx.y;
  int oy = noy % H;
  int n  = noy / H;
  float acc = b0[co];
#pragma unroll
  for (int ky = 0; ky < 3; ++ky) {
    int y = oy + ky - 1;
    if ((unsigned)y >= (unsigned)H) continue;
#pragma unroll
    for (int kx = 0; kx < 3; ++kx) {
      int xx = ox + kx - 1;
      if ((unsigned)xx >= (unsigned)W) continue;
      const float* px = xin + (((size_t)n * H + y) * W + xx) * 3;
      const float* pw = w0 + ((ky * 3 + kx) * 3) * 32 + co;
      acc += px[0] * pw[0] + px[1] * pw[32] + px[2] * pw[64];
    }
  }
  float r = acc > 0.0f ? acc : 0.0f;
  float z = r * bn0[co] + bn0[32 + co];
  out[((size_t)noy * W + ox) * 32 + co] = (z >= 0.0f) ? (int8_t)1 : (int8_t)-1;
}

template <int KH, int CI, int WH>
__device__ __forceinline__ void stage_tiles(const int8_t* __restrict__ act,
                                            const int8_t* __restrict__ wt,
                                            int kc, long long bm, int bn_,
                                            int* __restrict__ As4,
                                            uint32_t asBase, uint32_t bsBase,
                                            int tid) {
  constexpr int K    = KH * KH * CI;
  constexpr int Kpad = (K + 63) & ~63;
  constexpr int HW   = WH * WH;
  (void)Kpad; (void)HW;

  {
    int col  = tid >> 3;
    int boff = (tid & 7) << 3;
    async_ld_b64(bsBase + (uint32_t)(col * 64 + boff),
                 wt + (size_t)(bn_ + col) * Kpad + (kc << 6) + boff);
  }
  if constexpr (KH == 1) {
    if ((kc << 6) + 64 <= K) {
#pragma unroll
      for (int i = 0; i < 2; ++i) {
        int idx16 = tid + i * 128;
        int r     = idx16 >> 2;
        int boff  = (idx16 & 3) << 4;
        async_ld_b128(asBase + (uint32_t)(r * 64 + boff),
                      act + (size_t)(bm + r) * CI + (kc << 6) + boff);
      }
    } else {
#pragma unroll
      for (int i = 0; i < 8; ++i) {
        int idx = tid + i * 128;
        int r   = idx >> 4;
        int kk  = (idx & 15) << 2;
        int k   = (kc << 6) + kk;
        int val = 0;
        if (k < K) val = *(const int*)(act + (size_t)(bm + r) * CI + k);
        As4[idx] = val;
      }
    }
  } else {
#pragma unroll
    for (int i = 0; i < 8; ++i) {
      int idx = tid + i * 128;
      int r   = idx >> 4;
      int kk  = (idx & 15) << 2;
      int k   = (kc << 6) + kk;
      int val = 0;
      long long m = bm + r;
      if (k < K) {
        int cell = k / CI;
        int ci   = k - cell * CI;
        int ky   = cell / 3;
        int kx   = cell - ky * 3;
        int n    = (int)(m / HW);
        int rem  = (int)(m - (long long)n * HW);
        int oy   = rem / WH;
        int ox   = rem - oy * WH;
        int y    = oy + ky - 1;
        int x    = ox + kx - 1;
        if ((unsigned)y < (unsigned)WH && (unsigned)x < (unsigned)WH) {
          val = *(const int*)(act + (((size_t)n * WH + y) * WH + x) * CI + ci);
        }
      }
      As4[idx] = val;
    }
  }
}

template <int KH, int CI, int WH, bool OUT_I8>
__global__ void bgemm_conv_t(const int8_t* __restrict__ act,
                             const int8_t* __restrict__ wt,
                             const float* __restrict__ bnv,
                             int8_t* __restrict__ out_i8,
                             float*  __restrict__ out_f32,
                             int Co, int coBase, int CoTotal) {
  constexpr int K    = KH * KH * CI;
  constexpr int Kpad = (K + 63) & ~63;
  constexpr int nK   = Kpad >> 6;

  __shared__ int As4[2][64 * 16];
  __shared__ int Bs4[2][16 * 16];

  const int tid  = threadIdx.x;
  const int lane = tid & 31;
  const int wv   = tid >> 5;
  const int hi   = lane >> 4;
  const int lo16 = lane & 15;

  const long long bm = (long long)blockIdx.x * 64;
  const int bn_ = blockIdx.y * 16;

  const uint32_t asBase0 = (uint32_t)(uintptr_t)&As4[0][0];
  const uint32_t bsBase0 = (uint32_t)(uintptr_t)&Bs4[0][0];

  v8i acc = {0, 0, 0, 0, 0, 0, 0, 0};

  stage_tiles<KH, CI, WH>(act, wt, 0, bm, bn_, As4[0], asBase0, bsBase0, tid);

#pragma unroll
  for (int kc = 0; kc < nK; ++kc) {
    const int cur = kc & 1;
    const int nxt = cur ^ 1;
    wait_async0();
    __syncthreads();
    if (kc + 1 < nK) {
      stage_tiles<KH, CI, WH>(act, wt, kc + 1, bm, bn_, As4[nxt],
                              asBase0 + (uint32_t)(nxt * 4096),
                              bsBase0 + (uint32_t)(nxt * 1024), tid);
    }

    const int* Arow = &As4[cur][((wv << 4) + lo16) << 4];
    v8i a;
#pragma unroll
    for (int j = 0; j < 4; ++j) {
      int d = (j << 2) + (hi << 1);
      a[2 * j]     = Arow[d];
      a[2 * j + 1] = Arow[d + 1];
    }
    const int* Brow = &Bs4[cur][lo16 << 4];
    v8i b;
#pragma unroll
    for (int j = 0; j < 4; ++j) b[2 * j] = Brow[(j << 2) + (hi << 1)];
#pragma unroll
    for (int j = 0; j < 4; ++j) b[2 * j + 1] = Brow[(j << 2) + (hi << 1) + 1];

    acc = __builtin_amdgcn_wmma_i32_16x16x64_iu8(
         true, a,  true, b, acc,
         false,  false);
  }

  const int col  = bn_ + lo16;
  const float g  = bnv[col];
  const float bb = bnv[Co + col];
#pragma unroll
  for (int r = 0; r < 8; ++r) {
    long long m = bm + (wv << 4) + (hi << 3) + r;
    int c = acc[r];
    float z = (c > 0 ? (float)c : 0.0f) * g + bb;
    size_t o = (size_t)m * CoTotal + coBase + col;
    if constexpr (OUT_I8) out_i8[o]  = (z >= 0.0f) ? (int8_t)1 : (int8_t)-1;
    else                  out_f32[o] = z;
  }
}

template <int C, int H>
__global__ void pool2_i8_t(const int8_t* __restrict__ in, int8_t* __restrict__ out) {
  constexpr int Ho = H / 2, Wo = H / 2;
  int txc = blockIdx.x * blockDim.x + threadIdx.x;
  if (txc >= Wo * C) return;
  int c  = txc % C;
  int ox = txc / C;
  int noy = blockIdx.y;
  int oy = noy % Ho;
  int n  = noy / Ho;
  const int8_t* base = in + (((size_t)n * H + 2 * oy) * H + 2 * ox) * C + c;
  constexpr int rs = H * C;
  int8_t m0 = base[0], m1 = base[C], m2 = base[rs], m3 = base[rs + C];
  int8_t a = m0 > m1 ? m0 : m1;
  int8_t b = m2 > m3 ? m2 : m3;
  out[((size_t)noy * Wo + ox) * C + c] = a > b ? a : b;
}

__global__ void avgpool_kernel(const float* __restrict__ in, float* __restrict__ out) {
  int idx = blockIdx.x * blockDim.x + threadIdx.x;
  if (idx >= 32 * 64) return;
  int n = idx >> 6;
  int c = idx & 63;
  float s = 0.0f;
  for (int i = 0; i < 196; ++i) s += in[((size_t)n * 196 + i) * 64 + c];
  out[idx] = s * (1.0f / 196.0f);
}

__global__ void fc_kernel(const float* __restrict__ pooled, const float* __restrict__ wfc,
                          const float* __restrict__ bfc, float* __restrict__ out) {
  int idx = blockIdx.x * blockDim.x + threadIdx.x;
  if (idx >= 32 * 1000) return;
  int n = idx / 1000;
  int j = idx - n * 1000;
  float s = bfc[j];
#pragma unroll 8
  for (int k = 0; k < 64; ++k) s += pooled[n * 64 + k] * wfc[k * 1000 + j];
  out[idx] = s;
}

static inline int kpad64(int k) { return (k + 63) & ~63; }
static inline size_t align256(size_t v) { return (v + 255) & ~(size_t)255; }

extern "C" void kernel_launch(void* const* d_in, const int* in_sizes, int n_in,
                              void* d_out, int out_size, void* d_ws, size_t ws_size,
                              hipStream_t stream) {
  (void)in_sizes; (void)n_in; (void)out_size; (void)ws_size;
  const float* x   = (const float*)d_in[0];
  const float* w0  = (const float*)d_in[1];
  const float* b0  = (const float*)d_in[2];
  const float* bn0 = (const float*)d_in[3];
  const float* wfc = (const float*)d_in[34];
  const float* bfc = (const float*)d_in[35];
  float* out = (float*)d_out;

  int8_t* ws8 = (int8_t*)d_ws;
  const size_t szAB = 32ull * 224 * 224 * 64;
  const size_t szY  = 32ull * 224 * 224 * 32;
  size_t offA = 0;
  size_t offB = align256(offA + szAB);
  size_t offY = align256(offB + szAB);
  size_t woff = align256(offY + szY);

  int8_t* bufA = ws8 + offA;
  int8_t* bufB = ws8 + offB;
  int8_t* bufY = ws8 + offY;

  const int fireCi[5] = {32, 64, 96, 128, 96};
  const int sqCo[5]   = {32, 64, 64, 64, 32};
  const int exCo[5]   = {32, 48, 64, 48, 32};

  int8_t* wp[15];
  const float* bnp[15];
  for (int f = 0; f < 5; ++f) {
    int base = 4 + f * 6;
    const int KHs[3] = {1, 1, 3};
    const int Cis[3] = {fireCi[f], sqCo[f], sqCo[f]};
    const int Cos[3] = {sqCo[f], exCo[f], exCo[f]};
    for (int j = 0; j < 3; ++j) {
      int li = f * 3 + j;
      int KH = KHs[j], Ci = Cis[j], Co = Cos[j];
      int Kpad = kpad64(KH * KH * Ci);
      wp[li]  = ws8 + woff;
      bnp[li] = (const float*)d_in[base + 2 * j + 1];
      dim3 rg((unsigned)((Kpad + 255) / 256), (unsigned)Co);
      repack_w_kernel<<<rg, 256, 0, stream>>>(
          (const float*)d_in[base + 2 * j], wp[li], KH, KH, Ci, Co, Kpad);
      woff = align256(woff + (size_t)Co * Kpad);
    }
  }
  size_t offF = align256(woff);
  float* fbuf = (float*)(ws8 + offF);
  size_t offP = align256(offF + 32ull * 196 * 64 * 4);
  float* pooled = (float*)(ws8 + offP);

  stem_conv_kernel<<<dim3(28, 32 * 224), 256, 0, stream>>>(x, w0, b0, bn0, bufA);

  auto grd = [](int WH, int Co) {
    return dim3((unsigned)((32ll * WH * WH) / 64), (unsigned)(Co / 16));
  };

  bgemm_conv_t<1, 32, 224, true><<<grd(224, 32), 128, 0, stream>>>(bufA, wp[0],  bnp[0],  bufY, nullptr, 32, 0,  32);
  bgemm_conv_t<1, 32, 224, true><<<grd(224, 32), 128, 0, stream>>>(bufY, wp[1],  bnp[1],  bufB, nullptr, 32, 0,  64);
  bgemm_conv_t<3, 32, 224, true><<<grd(224, 32), 128, 0, stream>>>(bufY, wp[2],  bnp[2],  bufB, nullptr, 32, 32, 64);
  pool2_i8_t<64, 224><<<dim3((112 * 64 + 255) / 256, 32 * 112), 256, 0, stream>>>(bufB, bufA);

  bgemm_conv_t<1, 64, 112, true><<<grd(112, 64), 128, 0, stream>>>(bufA, wp[3],  bnp[3],  bufY, nullptr, 64, 0,  64);
  bgemm_conv_t<1, 64, 112, true><<<grd(112, 48), 128, 0, stream>>>(bufY, wp[4],  bnp[4],  bufB, nullptr, 48, 0,  96);
  bgemm_conv_t<3, 64, 112, true><<<grd(112, 48), 128, 0, stream>>>(bufY, wp[5],  bnp[5],  bufB, nullptr, 48, 48, 96);
  pool2_i8_t<96, 112><<<dim3((56 * 96 + 255) / 256, 32 * 56), 256, 0, stream>>>(bufB, bufA);

  bgemm_conv_t<1, 96, 56, true><<<grd(56, 64), 128, 0, stream>>>(bufA, wp[6],  bnp[6],  bufY, nullptr, 64, 0,  64);
  bgemm_conv_t<1, 64, 56, true><<<grd(56, 64), 128, 0, stream>>>(bufY, wp[7],  bnp[7],  bufB, nullptr, 64, 0,  128);
  bgemm_conv_t<3, 64, 56, true><<<grd(56, 64), 128, 0, stream>>>(bufY, wp[8],  bnp[8],  bufB, nullptr, 64, 64, 128);
  pool2_i8_t<128, 56><<<dim3((28 * 128 + 255) / 256, 32 * 28), 256, 0, stream>>>(bufB, bufA);

  bgemm_conv_t<1, 128, 28, true><<<grd(28, 64), 128, 0, stream>>>(bufA, wp[9],  bnp[9],  bufY, nullptr, 64, 0,  64);
  bgemm_conv_t<1, 64, 28, true><<<grd(28, 48), 128, 0, stream>>>(bufY, wp[10], bnp[10], bufB, nullptr, 48, 0,  96);
  bgemm_conv_t<3, 64, 28, true><<<grd(28, 48), 128, 0, stream>>>(bufY, wp[11], bnp[11], bufB, nullptr, 48, 48, 96);
  pool2_i8_t<96, 28><<<dim3((14 * 96 + 255) / 256, 32 * 14), 256, 0, stream>>>(bufB, bufA);

  bgemm_conv_t<1, 96, 14, true ><<<grd(14, 32), 128, 0, stream>>>(bufA, wp[12], bnp[12], bufY, nullptr, 32, 0,  32);
  bgemm_conv_t<1, 32, 14, false><<<grd(14, 32), 128, 0, stream>>>(bufY, wp[13], bnp[13], nullptr, fbuf, 32, 0,  64);
  bgemm_conv_t<3, 32, 14, false><<<grd(14, 32), 128, 0, stream>>>(bufY, wp[14], bnp[14], nullptr, fbuf, 32, 32, 64);

  avgpool_kernel<<<(32 * 64 + 255) / 256, 256, 0, stream>>>(fbuf, pooled);
  fc_kernel<<<(32 * 1000 + 255) / 256, 256, 0, stream>>>(pooled, wfc, bfc, out);
}
